// MoELayer_6116033429798
// MI455X (gfx1250) — compile-verified, matrix operand order checked by reading, not run on hardware
//
#include <hip/hip_runtime.h>
#include <hip/hip_bf16.h>

#define Hdim 1024
#define Ddim 2048
#define NE   8
#define TT   8192

typedef __attribute__((ext_vector_type(16))) __bf16 v16bf;
typedef __attribute__((ext_vector_type(8)))  __bf16 v8bf;
typedef __attribute__((ext_vector_type(8)))  float  v8f;

// async copy: global (64-bit addr) -> LDS (32-bit byte addr), 16B per lane.
// INST_OFFSET applies to both LDS and global addresses (ISA 08_async_tensor §4.4),
// so a whole row copy needs only one address pair + immediate offsets.
__device__ __forceinline__ void async_g2l_b128(unsigned ldsAddr, unsigned long long gAddr,
                                               int ofs) {
    asm volatile("global_load_async_to_lds_b128 %0, %1, off offset:%2"
                 :: "v"(ldsAddr), "v"(gAddr), "n"(ofs) : "memory");
}
__device__ __forceinline__ void wait_async0() {
    asm volatile("s_wait_asynccnt 0" ::: "memory");
}
__device__ __forceinline__ unsigned lds_addr_of(const void* p) {
    return (unsigned)(unsigned long long)(__attribute__((address_space(3))) const void*)p;
}

// ---------------- zero per-expert counters ----------------
__global__ void k_zero(int* cnt) {
    if (threadIdx.x < NE) cnt[threadIdx.x] = 0;
}

// ------- router: softmax(x@Wr) top-2; list entries = 2*token+pos; x->bf16 -------
__global__ __launch_bounds__(256) void k_router(const float* __restrict__ x,
                                                const float* __restrict__ Wr,
                                                __bf16* __restrict__ xb,
                                                int* __restrict__ cnt,
                                                int* __restrict__ tok,
                                                float* __restrict__ wt) {
    const int wid  = threadIdx.x >> 5;
    const int lane = threadIdx.x & 31;
    const int t    = blockIdx.x * 8 + wid;

    float acc[NE];
#pragma unroll
    for (int e = 0; e < NE; ++e) acc[e] = 0.f;

    const float* xr  = x  + (size_t)t * Hdim;
    __bf16*      xbr = xb + (size_t)t * Hdim;

    for (int k = lane; k < Hdim; k += 32) {
        float xv = xr[k];
        xbr[k] = (__bf16)xv;
        const float4* w = (const float4*)(Wr + (size_t)k * NE);
        float4 w0 = w[0], w1 = w[1];
        acc[0] += xv * w0.x; acc[1] += xv * w0.y;
        acc[2] += xv * w0.z; acc[3] += xv * w0.w;
        acc[4] += xv * w1.x; acc[5] += xv * w1.y;
        acc[6] += xv * w1.z; acc[7] += xv * w1.w;
    }
#pragma unroll
    for (int e = 0; e < NE; ++e)
        for (int off = 16; off; off >>= 1) acc[e] += __shfl_xor(acc[e], off, 32);

    if (lane == 0) {
        float m = acc[0];
#pragma unroll
        for (int e = 1; e < NE; ++e) m = fmaxf(m, acc[e]);
        float p[NE], Z = 0.f;
#pragma unroll
        for (int e = 0; e < NE; ++e) { p[e] = __expf(acc[e] - m); Z += p[e]; }
        int i0 = 0;
#pragma unroll
        for (int e = 1; e < NE; ++e) if (acc[e] > acc[i0]) i0 = e;
        int i1 = (i0 == 0) ? 1 : 0;
#pragma unroll
        for (int e = 0; e < NE; ++e) if (e != i0 && acc[e] > acc[i1]) i1 = e;
        float invZ = __builtin_amdgcn_rcpf(Z);
        float p0 = p[i0] * invZ, p1 = p[i1] * invZ;
        float inv = __builtin_amdgcn_rcpf(p0 + p1 + 1e-20f);
        int s0 = atomicAdd(&cnt[i0], 1);
        tok[i0 * TT + s0] = 2 * t + 0; wt[i0 * TT + s0] = p0 * inv;
        int s1 = atomicAdd(&cnt[i1], 1);
        tok[i1 * TT + s1] = 2 * t + 1; wt[i1 * TT + s1] = p1 * inv;
    }
}

// ---------------- transpose + convert fp32 [R][C] -> bf16 [C][R] ----------------
__global__ __launch_bounds__(256) void k_txc(const float* __restrict__ src,
                                             __bf16* __restrict__ dst,
                                             int R, int C) {
    __shared__ float tile[64][65];
    const size_t boff = (size_t)blockIdx.z * R * C;
    const float* s = src + boff;
    __bf16*      d = dst + boff;
    const int tr = blockIdx.y * 64, tc = blockIdx.x * 64;
    const int r = threadIdx.x >> 2;
    const int cs = (threadIdx.x & 3) * 16;

    const float* sp = s + (size_t)(tr + r) * C + tc + cs;
#pragma unroll
    for (int i = 0; i < 16; i += 4) {
        float4 v = *(const float4*)(sp + i);
        tile[r][cs + i + 0] = v.x; tile[r][cs + i + 1] = v.y;
        tile[r][cs + i + 2] = v.z; tile[r][cs + i + 3] = v.w;
    }
    __syncthreads();
    __bf16* dp = d + (size_t)(tc + r) * R + tr + cs;
    __bf16 buf[16];
#pragma unroll
    for (int i = 0; i < 16; ++i) buf[i] = (__bf16)tile[cs + i][r];
    *(v8bf*)(dp)     = *(v8bf*)(buf);
    *(v8bf*)(dp + 8) = *(v8bf*)(buf + 8);
}

// =============== kernel A: H1 = silu(X@Wg) * (X@Wu), 128 tokens/WG ===============
static constexpr int XS_STRIDE  = Hdim + 8;   // 1032
static constexpr int HST_STRIDE = 72;         // 64 + 8
static constexpr size_t SMEM_A =
    (size_t)(128 * XS_STRIDE + 128 * HST_STRIDE) * 2 + 128 * 8;

template <bool SHARED>
__global__ __launch_bounds__(256) void k_ffn1(const __bf16* __restrict__ xb,
                                              const __bf16* __restrict__ Wg,
                                              const __bf16* __restrict__ Wu,
                                              const int* __restrict__ cnt,
                                              const int* __restrict__ tok,
                                              __bf16* __restrict__ h1) {
    extern __shared__ char smem[];
    __bf16* Xs  = (__bf16*)smem;                    // [128][XS_STRIDE]
    __bf16* Hst = Xs + 128 * XS_STRIDE;             // [128][HST_STRIDE]
    int* rowSrc = (int*)(Hst + 128 * HST_STRIDE);
    int* rowDst = rowSrc + 128;

    int expert, tile, nTok;
    if (SHARED) { expert = 0; tile = blockIdx.x; nTok = TT; }
    else {
        expert = blockIdx.x >> 6;
        tile   = blockIdx.x & 63;
        nTok   = cnt[expert];
        if (tile * 128 >= nTok) return;
    }
    const __bf16* Wg_e = SHARED ? Wg : Wg + (size_t)expert * Hdim * Ddim;
    const __bf16* Wu_e = SHARED ? Wu : Wu + (size_t)expert * Hdim * Ddim;

    const int tid = threadIdx.x, wid = tid >> 5, lane = tid & 31;
    const int half = lane >> 4, lm = lane & 15;

    if (tid < 128) {
        int gRow = tile * 128 + tid;
        if (SHARED)           { rowSrc[tid] = gRow; rowDst[tid] = gRow; }
        else if (gRow < nTok) { int tc = tok[expert * TT + gRow];
                                rowSrc[tid] = tc >> 1; rowDst[tid] = tc; }
        else                  { rowSrc[tid] = -1; rowDst[tid] = -1; }
    }
    __syncthreads();
    {   // gather X via async DMA burst: 2 threads per row, 512 bf16 each
        int r = tid >> 1, s = (tid & 1) * 512;
        int t = rowSrc[r];
        __bf16* dst = Xs + r * XS_STRIDE + s;
        if (t >= 0) {
            unsigned la = lds_addr_of(dst);
            unsigned long long ga = (unsigned long long)(xb + (size_t)t * Hdim + s);
#pragma unroll
            for (int i = 0; i < 64; ++i)
                async_g2l_b128(la, ga, i * 16);
        } else {
            v8bf z = {};
#pragma unroll
            for (int i = 0; i < 512; i += 8) *(v8bf*)(dst + i) = z;
        }
    }
    wait_async0();
    __syncthreads();

    const v8f zero8 = {0.f,0.f,0.f,0.f,0.f,0.f,0.f,0.f};
    const __bf16* aBase = Xs + (wid * 16 + lm) * XS_STRIDE + half * 8;

    for (int nb = 0; nb < Ddim; nb += 64) {
        v8f accG[4], accU[4];
        const __bf16* bg[4];
        const __bf16* bu[4];
#pragma unroll
        for (int na = 0; na < 4; ++na) {
            accG[na] = zero8; accU[na] = zero8;
            int n = nb + na * 16 + lm;
            bg[na] = Wg_e + (size_t)n * Hdim + half * 8;
            bu[na] = Wu_e + (size_t)n * Hdim + half * 8;
        }
        for (int k0 = 0; k0 < Hdim; k0 += 32) {
            const __bf16* ap = aBase + k0;
            v8bf a0 = *(const v8bf*)ap, a1 = *(const v8bf*)(ap + 16);
            v16bf a = __builtin_shufflevector(a0, a1, 0,1,2,3,4,5,6,7,8,9,10,11,12,13,14,15);
#pragma unroll
            for (int na = 0; na < 4; ++na) {
                __builtin_prefetch(bg[na] + k0 + 256, 0, 1);
                v8bf g0 = *(const v8bf*)(bg[na] + k0), g1 = *(const v8bf*)(bg[na] + k0 + 16);
                v16bf bG = __builtin_shufflevector(g0, g1, 0,1,2,3,4,5,6,7,8,9,10,11,12,13,14,15);
                accG[na] = __builtin_amdgcn_wmma_f32_16x16x32_bf16(false, a, false, bG,
                                                                   (short)0, accG[na], false, false);
                v8bf u0 = *(const v8bf*)(bu[na] + k0), u1 = *(const v8bf*)(bu[na] + k0 + 16);
                v16bf bU = __builtin_shufflevector(u0, u1, 0,1,2,3,4,5,6,7,8,9,10,11,12,13,14,15);
                accU[na] = __builtin_amdgcn_wmma_f32_16x16x32_bf16(false, a, false, bU,
                                                                   (short)0, accU[na], false, false);
            }
        }
        __syncthreads();   // previous Hst copy complete
#pragma unroll
        for (int na = 0; na < 4; ++na) {
#pragma unroll
            for (int j = 0; j < 8; ++j) {
                int r = wid * 16 + half * 8 + j;
                int c = na * 16 + lm;
                float g = accG[na][j], u = accU[na][j];
                float h = g * __builtin_amdgcn_rcpf(1.f + __expf(-g)) * u;
                Hst[r * HST_STRIDE + c] = (__bf16)h;
            }
        }
        __syncthreads();
        {   // coalesced copy Hst -> h1
            int r = tid >> 1, s = (tid & 1) * 32;
            int d = rowDst[r];
            if (d >= 0) {
                __bf16* dp = h1 + (size_t)d * Ddim + nb + s;
                const __bf16* sp = Hst + r * HST_STRIDE + s;
#pragma unroll
                for (int i = 0; i < 32; i += 8) *(v8bf*)(dp + i) = *(const v8bf*)(sp + i);
            }
        }
    }
}

// =============== kernel B: Out += w * (H1 @ Wd), 64 tokens/WG ===============
static constexpr int HS_STRIDE = Ddim + 8;   // 2056
static constexpr size_t SMEM_B = (size_t)(64 * HS_STRIDE) * 2 + 64 * 12;

template <bool SHARED>
__global__ __launch_bounds__(256) void k_ffn2(const __bf16* __restrict__ h1,
                                              const __bf16* __restrict__ Wd,
                                              const int* __restrict__ cnt,
                                              const int* __restrict__ tok,
                                              const float* __restrict__ wt,
                                              float* __restrict__ out) {
    extern __shared__ char smem[];
    __bf16* Hs  = (__bf16*)smem;                 // [64][HS_STRIDE]
    int* srcRow = (int*)(Hs + 64 * HS_STRIDE);
    int* outTok = srcRow + 64;
    float* tokW = (float*)(outTok + 64);

    int expert, tile, nTok;
    if (SHARED) { expert = 0; tile = blockIdx.x; nTok = TT; }
    else {
        expert = blockIdx.x >> 7;
        tile   = blockIdx.x & 127;
        nTok   = cnt[expert];
        if (tile * 64 >= nTok) return;
    }
    const __bf16* Wd_e = SHARED ? Wd : Wd + (size_t)expert * Ddim * Hdim;

    const int tid = threadIdx.x, wid = tid >> 5, lane = tid & 31;
    const int wr = wid >> 1, wc = wid & 1;
    const int half = lane >> 4, lm = lane & 15;

    if (tid < 64) {
        int gRow = tile * 64 + tid;
        if (SHARED)           { srcRow[tid] = gRow; outTok[tid] = gRow; tokW[tid] = 1.f; }
        else if (gRow < nTok) { int tc = tok[expert * TT + gRow];
                                srcRow[tid] = tc; outTok[tid] = tc >> 1;
                                tokW[tid] = wt[expert * TT + gRow]; }
        else                  { srcRow[tid] = -1; outTok[tid] = -1; tokW[tid] = 0.f; }
    }
    __syncthreads();
    {   // gather H1 via async DMA burst: 4 threads per row, 512 bf16 each
        int r = tid >> 2, s = (tid & 3) * 512;
        int t = srcRow[r];
        __bf16* dst = Hs + r * HS_STRIDE + s;
        if (t >= 0) {
            unsigned la = lds_addr_of(dst);
            unsigned long long ga = (unsigned long long)(h1 + (size_t)t * Ddim + s);
#pragma unroll
            for (int i = 0; i < 64; ++i)
                async_g2l_b128(la, ga, i * 16);
        } else {
            v8bf z = {};
#pragma unroll
            for (int i = 0; i < 512; i += 8) *(v8bf*)(dst + i) = z;
        }
    }
    wait_async0();
    __syncthreads();

    const v8f zero8 = {0.f,0.f,0.f,0.f,0.f,0.f,0.f,0.f};
    const __bf16* aBase = Hs + (16 * wr + lm) * HS_STRIDE + half * 8;

    for (int nb = 0; nb < Hdim; nb += 128) {
        v8f acc[4];
        const __bf16* bd[4];
#pragma unroll
        for (int na = 0; na < 4; ++na) {
            acc[na] = zero8;
            int n = nb + (wc * 4 + na) * 16 + lm;
            bd[na] = Wd_e + (size_t)n * Ddim + half * 8;
        }
        for (int k0 = 0; k0 < Ddim; k0 += 32) {
            const __bf16* ap = aBase + k0;
            v8bf a0 = *(const v8bf*)ap, a1 = *(const v8bf*)(ap + 16);
            v16bf a = __builtin_shufflevector(a0, a1, 0,1,2,3,4,5,6,7,8,9,10,11,12,13,14,15);
#pragma unroll
            for (int na = 0; na < 4; ++na) {
                __builtin_prefetch(bd[na] + k0 + 256, 0, 1);
                v8bf b0 = *(const v8bf*)(bd[na] + k0), b1 = *(const v8bf*)(bd[na] + k0 + 16);
                v16bf b = __builtin_shufflevector(b0, b1, 0,1,2,3,4,5,6,7,8,9,10,11,12,13,14,15);
                acc[na] = __builtin_amdgcn_wmma_f32_16x16x32_bf16(false, a, false, b,
                                                                  (short)0, acc[na], false, false);
            }
        }
#pragma unroll
        for (int na = 0; na < 4; ++na) {
            int n = nb + (wc * 4 + na) * 16 + lm;
#pragma unroll
            for (int j = 0; j < 8; ++j) {
                int r = 16 * wr + half * 8 + j;
                int t = outTok[r];
                if (SHARED) {
                    out[(size_t)t * Hdim + n] = acc[na][j];
                } else if (t >= 0) {
                    atomicAdd(out + (size_t)t * Hdim + n, tokW[r] * acc[na][j]);
                }
            }
        }
    }
}

extern "C" void kernel_launch(void* const* d_in, const int* in_sizes, int n_in,
                              void* d_out, int out_size, void* d_ws, size_t ws_size,
                              hipStream_t stream) {
    const float* x  = (const float*)d_in[0];
    const float* Wr = (const float*)d_in[1];
    const float* Wg = (const float*)d_in[2];
    const float* Wu = (const float*)d_in[3];
    const float* Wd = (const float*)d_in[4];
    const float* Sg = (const float*)d_in[5];
    const float* Su = (const float*)d_in[6];
    const float* Sd = (const float*)d_in[7];
    float* out = (float*)d_out;

    char* ws = (char*)d_ws;
    const size_t MB = 1024 * 1024;
    int*    cnt = (int*)ws;
    int*    tok = (int*)(ws + 256);
    float*  wt  = (float*)(ws + 256 + NE * TT * 4);
    __bf16* xb  = (__bf16*)(ws + 1 * MB);      // 16 MB
    __bf16* Wgb = (__bf16*)(ws + 17 * MB);     // 32 MB  [e][D][H]
    __bf16* Wub = (__bf16*)(ws + 49 * MB);     // 32 MB  [e][D][H]
    __bf16* Wdb = (__bf16*)(ws + 81 * MB);     // 32 MB  [e][H][D]
    __bf16* Sgb = (__bf16*)(ws + 113 * MB);    // 4 MB
    __bf16* Sub = (__bf16*)(ws + 117 * MB);    // 4 MB
    __bf16* Sdb = (__bf16*)(ws + 121 * MB);    // 4 MB
    __bf16* h1r = (__bf16*)(ws + 126 * MB);    // 64 MB  [2*T][D]
    __bf16* h1s = (__bf16*)(ws + 190 * MB);    // 32 MB  [T][D]

    k_zero<<<1, 32, 0, stream>>>(cnt);
    k_router<<<TT / 8, 256, 0, stream>>>(x, Wr, xb, cnt, tok, wt);

    k_txc<<<dim3(Ddim/64, Hdim/64, NE), 256, 0, stream>>>(Wg, Wgb, Hdim, Ddim);
    k_txc<<<dim3(Ddim/64, Hdim/64, NE), 256, 0, stream>>>(Wu, Wub, Hdim, Ddim);
    k_txc<<<dim3(Hdim/64, Ddim/64, NE), 256, 0, stream>>>(Wd, Wdb, Ddim, Hdim);
    k_txc<<<dim3(Ddim/64, Hdim/64, 1),  256, 0, stream>>>(Sg, Sgb, Hdim, Ddim);
    k_txc<<<dim3(Ddim/64, Hdim/64, 1),  256, 0, stream>>>(Su, Sub, Hdim, Ddim);
    k_txc<<<dim3(Hdim/64, Ddim/64, 1),  256, 0, stream>>>(Sd, Sdb, Ddim, Hdim);

    // GEMM1 (+silu*mul) -> H1
    k_ffn1<true><<<TT / 128, 256, SMEM_A, stream>>>(xb, Sgb, Sub, nullptr, nullptr, h1s);
    k_ffn1<false><<<NE * 64, 256, SMEM_A, stream>>>(xb, Wgb, Wub, cnt, tok, h1r);

    // GEMM2 -> out (shared writes, routed atomically accumulates)
    k_ffn2<true><<<TT / 64, 256, SMEM_B, stream>>>(h1s, Sdb, nullptr, nullptr, nullptr, out);
    k_ffn2<false><<<NE * 128, 256, SMEM_B, stream>>>(h1r, Wdb, cnt, tok, wt, out);
}
